// MMKGBase_30391188587216
// MI455X (gfx1250) — matrix operand order repaired and checked by reading, not compiled, not run on hardware
//
#include <hip/hip_runtime.h>
#include <stdint.h>

// ---------------- problem constants (from reference setup_inputs) ----------
#define NENT   100000   // entities
#define DIM    200      // embedding dim
#define KMOD   768      // modality raw dim
#define NREL2  400      // 2*num_rel
#define EDIR   200000   // edges per direction (E2/2)
#define NBATCH 2048     // query batch

// ---------------- WMMA types ----------------------------------------------
typedef __attribute__((ext_vector_type(16))) __bf16 bf16x16;
typedef __attribute__((ext_vector_type(8)))  float  f32x8;

struct __attribute__((aligned(16))) U4 { unsigned int a, b, c, d; };
union Frag { bf16x16 v; U4 q[2]; };

// round-half-up fp32->bf16 (ties differ from RNE; bf16 is approximate anyway)
__device__ __forceinline__ unsigned short f32_to_bf16_bits(float f) {
  return (unsigned short)((__float_as_uint(f) + 0x8000u) >> 16);
}
// pack two bf16 into one dword with a single v_perm_b32:
// D = {hi16(a+0x8000), hi16(b+0x8000)}; pool = {S0=ua (idx 4-7), S1=ub (idx 0-3)}
__device__ __forceinline__ uint32_t pack_bf16x2(float a, float b) {
  uint32_t ua = __float_as_uint(a) + 0x8000u;
  uint32_t ub = __float_as_uint(b) + 0x8000u;
  return __builtin_amdgcn_perm(ua, ub, 0x07060302u);
}

// ---------------- GEMM: C[M,N] = (A (+Aadd)) @ B, bf16 WMMA, f32 accum -----
// Block: 256 threads = 8 waves; block tile 128x64; wave w owns rows
// [w*16, w*16+16) x 64 cols (4 WMMA accumulators). K staged in 32-chunks.
// Double-buffered LDS (one barrier per chunk) + register prefetch of the
// next chunk so global loads overlap the WMMAs. LDS rows padded to 48
// ushorts = 96B so fragment reads are 16B-aligned ds_load_b128.
template <bool HAS_ADD>
__global__ __launch_bounds__(256) void gemm_bf16_wmma_kernel(
    const float* __restrict__ A, const float* __restrict__ Aadd,
    const float* __restrict__ B, float* __restrict__ C,
    int M, int N, int K, int accumulate)
{
  constexpr int TM = 128, TN = 64, TK = 32;
  constexpr int LDSS = 48;
  __shared__ __attribute__((aligned(16))) unsigned short As[2][TM * LDSS];
  __shared__ __attribute__((aligned(16))) unsigned short Bs[2][TN * LDSS];

  const int tid  = threadIdx.x;
  const int wave = tid >> 5;      // 0..7
  const int lane = tid & 31;
  const int half = lane >> 4;     // 0: lanes 0-15, 1: lanes 16-31
  const int l15  = lane & 15;
  const int rowBlk = blockIdx.x * TM;
  const int colBlk = blockIdx.y * TN;

  const f32x8 zero = {0.f,0.f,0.f,0.f,0.f,0.f,0.f,0.f};
  f32x8 acc[4] = {zero, zero, zero, zero};

  const bool aRowsFull = (rowBlk + TM <= M);
  const bool bColsFull = (colBlk + TN <= N);

  // staging registers (unified layout for fast / ragged paths)
  float4 ra[4], rad[4], rb[2];

  auto loadChunk = [&](int k0) {
    const bool kFull = (k0 + TK <= K);
    const bool fastA = aRowsFull && kFull;   // uniform
    const bool fastB = bColsFull && kFull;   // uniform
#pragma unroll
    for (int i = 0; i < 4; ++i) {            // A: 128x32, 16 floats/thread
      int idx = i * 256 + tid;               // float4 index 0..1023
      int r   = idx >> 3;                    // 8 vec4 per 32-wide row
      int k4  = (idx & 7) << 2;
      int gr  = rowBlk + r;
      if (fastA) {
        size_t off = (size_t)gr * K + (k0 + k4);
        ra[i] = *(const float4*)(A + off);
        if (HAS_ADD) rad[i] = *(const float4*)(Aadd + off);
      } else {
        float t[4];
#pragma unroll
        for (int c = 0; c < 4; ++c) {
          int gk = k0 + k4 + c;
          bool ok = (gr < M) && (gk < K);
          size_t off = ok ? ((size_t)gr * K + gk) : 0;   // clamped, always load
          float v = A[off];
          if (HAS_ADD) v += Aadd[off];
          t[c] = ok ? v : 0.f;
        }
        ra[i] = make_float4(t[0], t[1], t[2], t[3]);
        if (HAS_ADD) rad[i] = make_float4(0.f, 0.f, 0.f, 0.f);
      }
    }
#pragma unroll
    for (int i = 0; i < 2; ++i) {            // B: 32x64, 8 floats/thread
      int idx = i * 256 + tid;               // float4 index 0..511
      int kk  = idx >> 4;                    // 16 vec4 per 64-wide row
      int c4  = (idx & 15) << 2;
      int gk  = k0 + kk;
      if (fastB) {
        rb[i] = *(const float4*)(B + (size_t)gk * N + (colBlk + c4));
      } else {
        float t[4];
#pragma unroll
        for (int c = 0; c < 4; ++c) {
          int gc = colBlk + c4 + c;
          bool ok = (gk < K) && (gc < N);
          size_t off = ok ? ((size_t)gk * N + gc) : 0;
          float v = B[off];
          t[c] = ok ? v : 0.f;
        }
        rb[i] = make_float4(t[0], t[1], t[2], t[3]);
      }
    }
  };

  auto storeChunk = [&](int p) {
#pragma unroll
    for (int i = 0; i < 4; ++i) {
      int idx = i * 256 + tid;
      int r   = idx >> 3;
      int k4  = (idx & 7) << 2;
      float4 v = ra[i];
      if (HAS_ADD) { v.x += rad[i].x; v.y += rad[i].y; v.z += rad[i].z; v.w += rad[i].w; }
      uint2 pk;
      pk.x = pack_bf16x2(v.y, v.x);   // dword = {hi: y, lo: x} -> K order x,y
      pk.y = pack_bf16x2(v.w, v.z);
      *(uint2*)&As[p][r * LDSS + k4] = pk;
    }
#pragma unroll
    for (int i = 0; i < 2; ++i) {            // transposed: Bs[col][k]
      int idx = i * 256 + tid;
      int kk  = idx >> 4;
      int c4  = (idx & 15) << 2;
      float4 v = rb[i];
      Bs[p][(c4 + 0) * LDSS + kk] = f32_to_bf16_bits(v.x);
      Bs[p][(c4 + 1) * LDSS + kk] = f32_to_bf16_bits(v.y);
      Bs[p][(c4 + 2) * LDSS + kk] = f32_to_bf16_bits(v.z);
      Bs[p][(c4 + 3) * LDSS + kk] = f32_to_bf16_bits(v.w);
    }
  };

  const int nChunks = (K + TK - 1) / TK;
  const int arow = wave * 16 + l15;
  const int kbA  = half * 16;      // A frag: K {kbA..kbA+15}
  const int kbB  = half * 16;      // B frag: K {kbB..kbB+15}

  loadChunk(0);
  int p = 0;
  for (int c = 0; c < nChunks; ++c) {
    storeChunk(p);
    __syncthreads();
    if (c + 1 < nChunks) loadChunk((c + 1) * TK);   // overlap with WMMAs below

    Frag af;
    af.q[0] = *(const U4*)&As[p][arow * LDSS + kbA];
    af.q[1] = *(const U4*)&As[p][arow * LDSS + kbA + 8];
    Frag bf0, bf1, bf2, bf3;
    bf0.q[0] = *(const U4*)&Bs[p][( 0 + l15) * LDSS + kbB];
    bf0.q[1] = *(const U4*)&Bs[p][( 0 + l15) * LDSS + kbB + 8];
    bf1.q[0] = *(const U4*)&Bs[p][(16 + l15) * LDSS + kbB];
    bf1.q[1] = *(const U4*)&Bs[p][(16 + l15) * LDSS + kbB + 8];
    bf2.q[0] = *(const U4*)&Bs[p][(32 + l15) * LDSS + kbB];
    bf2.q[1] = *(const U4*)&Bs[p][(32 + l15) * LDSS + kbB + 8];
    bf3.q[0] = *(const U4*)&Bs[p][(48 + l15) * LDSS + kbB];
    bf3.q[1] = *(const U4*)&Bs[p][(48 + l15) * LDSS + kbB + 8];

    acc[0] = __builtin_amdgcn_wmma_f32_16x16x32_bf16(
        false, af.v, false, bf0.v, (short)0, acc[0], false, false);
    acc[1] = __builtin_amdgcn_wmma_f32_16x16x32_bf16(
        false, af.v, false, bf1.v, (short)0, acc[1], false, false);
    acc[2] = __builtin_amdgcn_wmma_f32_16x16x32_bf16(
        false, af.v, false, bf2.v, (short)0, acc[2], false, false);
    acc[3] = __builtin_amdgcn_wmma_f32_16x16x32_bf16(
        false, af.v, false, bf3.v, (short)0, acc[3], false, false);
    p ^= 1;   // double buffer: safe to rewrite old buffer after next barrier
  }

  // C/D layout: vector elem r, lanes 0-15 -> M=r, lanes 16-31 -> M=r+8; N=lane&15
  const int mBase = rowBlk + wave * 16 + half * 8;
#pragma unroll
  for (int t = 0; t < 4; ++t) {
    int gc = colBlk + t * 16 + l15;
    if (gc < N) {
#pragma unroll
      for (int r = 0; r < 8; ++r) {
        int gr = mBase + r;
        if (gr < M) {
          size_t off = (size_t)gr * N + gc;
          float v = acc[t][r];
          C[off] = accumulate ? (C[off] + v) : v;
        }
      }
    }
  }
}

// ---------------- helpers ---------------------------------------------------
__global__ void zero_f32_kernel(float* __restrict__ p, int n) {
  int g = blockIdx.x * blockDim.x + threadIdx.x;
  if (g < n) p[g] = 0.f;
}

__global__ void deg_count_kernel(const int* __restrict__ src,
                                 float* __restrict__ deg, int E) {
  int g = blockIdx.x * blockDim.x + threadIdx.x;
  if (g < E) atomicAdd(&deg[src[g]], 1.0f);
}

// norm[e] = dinv(deg[src]) * dinv(deg[dst]); computed once per direction,
// shared by both layers (graph is static)
__global__ void edge_norm_kernel(const int* __restrict__ src,
                                 const int* __restrict__ dst,
                                 const float* __restrict__ deg,
                                 float* __restrict__ norm, int E) {
  int g = blockIdx.x * blockDim.x + threadIdx.x;
  if (g < E) {
    float ds = deg[src[g]], dt = deg[dst[g]];
    norm[g] = (ds > 0.f ? rsqrtf(ds) : 0.f) * (dt > 0.f ? rsqrtf(dt) : 0.f);
  }
}

// agg[dst] += norm[e] * (HW[src] - RW[etype]); one thread per (edge, 4 cols)
__global__ __launch_bounds__(256) void edge_scatter_kernel(
    const int* __restrict__ src, const int* __restrict__ dst,
    const int* __restrict__ etype, const float* __restrict__ norm,
    const float* __restrict__ HW, const float* __restrict__ RW,
    float* __restrict__ agg, int E)
{
  const int J4 = DIM / 4;                           // 50
  int g = blockIdx.x * blockDim.x + threadIdx.x;
  if (g >= E * J4) return;
  int e = g / J4;
  int j = (g - e * J4) * 4;
  int s = src[e], t = dst[e], et = etype[e];
  float nm = norm[e];
  const float4 h4 = *(const float4*)(HW + (size_t)s  * DIM + j);
  const float4 r4 = *(const float4*)(RW + (size_t)et * DIM + j);
  float* out = agg + (size_t)t * DIM + j;
  atomicAdd(out + 0, nm * (h4.x - r4.x));
  atomicAdd(out + 1, nm * (h4.y - r4.y));
  atomicAdd(out + 2, nm * (h4.z - r4.z));
  atomicAdd(out + 3, nm * (h4.w - r4.w));
}

// lrw[j] = loop_rel @ W  (1x200 @ 200x200)
__global__ void vecmat200_kernel(const float* __restrict__ v,
                                 const float* __restrict__ Wm,
                                 float* __restrict__ out) {
  int j = threadIdx.x;
  if (j < DIM) {
    float s = 0.f;
    for (int k = 0; k < DIM; ++k) s += v[k] * Wm[k * DIM + j];
    out[j] = s;
  }
}

// out_pre = (agg - lrw)/3 + b (in place) + per-column sum/sumsq via LDS atomics
__global__ __launch_bounds__(256) void post_stats_kernel(
    float* __restrict__ agg, const float* __restrict__ lrw,
    const float* __restrict__ bias, float* __restrict__ colsum,
    float* __restrict__ colsq, int Nrows)
{
  __shared__ float ssum[DIM];
  __shared__ float ssq[DIM];
  int tid = threadIdx.x;
  for (int j = tid; j < DIM; j += 256) { ssum[j] = 0.f; ssq[j] = 0.f; }
  __syncthreads();
  int rowBase = blockIdx.x * 128;
  for (int idx = tid; idx < 128 * DIM; idx += 256) {
    int r = rowBase + idx / DIM;
    if (r >= Nrows) break;                       // monotone per thread
    int j = idx - (idx / DIM) * DIM;
    size_t off = (size_t)r * DIM + j;
    float val = (agg[off] - lrw[j]) * (1.0f / 3.0f) + bias[j];
    agg[off] = val;
    atomicAdd(&ssum[j], val);                    // ds_add_f32
    atomicAdd(&ssq[j], val * val);
  }
  __syncthreads();
  for (int j = tid; j < DIM; j += 256) {
    atomicAdd(&colsum[j], ssum[j]);
    atomicAdd(&colsq[j],  ssq[j]);
  }
}

__global__ void bn_finalize_kernel(const float* __restrict__ colsum,
                                   const float* __restrict__ colsq,
                                   float* __restrict__ mu,
                                   float* __restrict__ rstd, float n) {
  int j = blockIdx.x * blockDim.x + threadIdx.x;
  if (j < DIM) {
    float m = colsum[j] / n;
    float var = colsq[j] / n - m * m;
    mu[j] = m;
    rstd[j] = rsqrtf(var + 1e-5f);
  }
}

__global__ void bn_tanh_kernel(const float* __restrict__ pre,
                               const float* __restrict__ mu,
                               const float* __restrict__ rstd,
                               float* __restrict__ out, size_t total) {
  size_t g = (size_t)blockIdx.x * blockDim.x + threadIdx.x;
  if (g < total) {
    int j = (int)(g % DIM);
    out[g] = tanhf((pre[g] - mu[j]) * rstd[j]);
  }
}

__global__ void gather_kernel(const int* __restrict__ sub,
                              const int* __restrict__ rel,
                              const float* __restrict__ xf,
                              const float* __restrict__ rf,
                              float* __restrict__ sube,
                              float* __restrict__ rele, int Bn) {
  int g = blockIdx.x * blockDim.x + threadIdx.x;
  if (g < Bn * DIM) {
    int i = g / DIM, j = g - i * DIM;
    sube[g] = xf[(size_t)sub[i] * DIM + j];
    rele[g] = rf[(size_t)rel[i] * DIM + j];
  }
}

// ---------------- host orchestration ---------------------------------------
extern "C" void kernel_launch(void* const* d_in, const int* in_sizes, int n_in,
                              void* d_out, int out_size, void* d_ws, size_t ws_size,
                              hipStream_t stream) {
  (void)in_sizes; (void)n_in; (void)out_size; (void)ws_size;

  const int*   sub        = (const int*)d_in[0];
  const int*   rel        = (const int*)d_in[1];
  const int*   edge_index = (const int*)d_in[2];   // [2, 2*EDIR] flat
  const int*   edge_type  = (const int*)d_in[3];
  const float* init_embed = (const float*)d_in[4];
  const float* img_raw    = (const float*)d_in[5];
  const float* img_trans  = (const float*)d_in[6];
  const float* desc_raw   = (const float*)d_in[7];
  const float* desc_trans = (const float*)d_in[8];
  const float* init_rel   = (const float*)d_in[9];
  const float* W[2][4]; const float* looprel[2]; const float* bias[2];
  for (int L = 0; L < 2; ++L) {
    int base = 10 + L * 6;
    W[L][0] = (const float*)d_in[base + 0];   // w_in
    W[L][1] = (const float*)d_in[base + 1];   // w_out
    W[L][2] = (const float*)d_in[base + 2];   // w_loop
    W[L][3] = (const float*)d_in[base + 3];   // w_rel
    looprel[L] = (const float*)d_in[base + 4];
    bias[L]    = (const float*)d_in[base + 5];
  }

  // workspace layout (floats), ~324 MB
  float* ws = (float*)d_ws;
  size_t o = 0;
  float* proj    = ws + o; o += (size_t)NENT * DIM;   // img+desc projection
  float* HW      = ws + o; o += (size_t)NENT * DIM;   // h @ w_in / w_out (reused)
  float* agg     = ws + o; o += (size_t)NENT * DIM;   // loop + scattered messages
  float* x1      = ws + o; o += (size_t)NENT * DIM;   // layer-1 node output
  float* degin   = ws + o; o += NENT;
  float* degout  = ws + o; o += NENT;
  float* normin  = ws + o; o += EDIR;
  float* normout = ws + o; o += EDIR;
  float* RW      = ws + o; o += (size_t)NREL2 * DIM;  // rel @ w (reused)
  float* r1      = ws + o; o += (size_t)NREL2 * DIM;
  float* r2      = ws + o; o += (size_t)NREL2 * DIM;
  float* lrw     = ws + o; o += 256;
  float* colsum  = ws + o; o += 256;
  float* colsq   = ws + o; o += 256;
  float* mu      = ws + o; o += 256;
  float* rstd    = ws + o; o += 256;

  float* sub_emb = (float*)d_out;
  float* rel_emb = sub_emb + (size_t)NBATCH * DIM;
  float* x2      = rel_emb + (size_t)NBATCH * DIM;   // [NENT, DIM] lives in d_out

  auto gemm = [&](const float* A, const float* Aadd, const float* B, float* C,
                  int M, int N, int K, int acc) {
    dim3 g((unsigned)((M + 127) / 128), (unsigned)((N + 63) / 64));
    if (Aadd)
      gemm_bf16_wmma_kernel<true><<<g, 256, 0, stream>>>(A, Aadd, B, C, M, N, K, acc);
    else
      gemm_bf16_wmma_kernel<false><<<g, 256, 0, stream>>>(A, nullptr, B, C, M, N, K, acc);
  };

  // modality projections: proj = img_raw@img_trans + desc_raw@desc_trans
  gemm(img_raw,  nullptr, img_trans,  proj, NENT, DIM, KMOD, 0);
  gemm(desc_raw, nullptr, desc_trans, proj, NENT, DIM, KMOD, 1);

  // degrees + per-edge norms (shared by both layers)
  zero_f32_kernel<<<(2 * NENT + 255) / 256, 256, 0, stream>>>(degin, 2 * NENT);
  deg_count_kernel<<<(EDIR + 255) / 256, 256, 0, stream>>>(edge_index, degin, EDIR);
  deg_count_kernel<<<(EDIR + 255) / 256, 256, 0, stream>>>(edge_index + EDIR, degout, EDIR);
  edge_norm_kernel<<<(EDIR + 255) / 256, 256, 0, stream>>>(
      edge_index, edge_index + 2 * EDIR, degin, normin, EDIR);
  edge_norm_kernel<<<(EDIR + 255) / 256, 256, 0, stream>>>(
      edge_index + EDIR, edge_index + 3 * EDIR, degout, normout, EDIR);

  const float* xin = init_embed;
  const float* rin = init_rel;
  for (int L = 0; L < 2; ++L) {
    float* xout = (L == 0) ? x1 : x2;
    float* rout = (L == 0) ? r1 : r2;

    // agg <- (x + proj) @ w_loop ; lrw <- loop_rel @ w_loop
    gemm(xin, proj, W[L][2], agg, NENT, DIM, DIM, 0);
    vecmat200_kernel<<<1, 256, 0, stream>>>(looprel[L], W[L][2], lrw);

    // in-direction: HW <- h@w_in, RW <- rel@w_in, scatter
    gemm(xin, proj, W[L][0], HW, NENT, DIM, DIM, 0);
    gemm(rin, nullptr, W[L][0], RW, NREL2, DIM, DIM, 0);
    edge_scatter_kernel<<<(EDIR * (DIM / 4) + 255) / 256, 256, 0, stream>>>(
        edge_index, edge_index + 2 * EDIR, edge_type, normin, HW, RW, agg, EDIR);

    // out-direction: HW <- h@w_out, RW <- rel@w_out, scatter
    gemm(xin, proj, W[L][1], HW, NENT, DIM, DIM, 0);
    gemm(rin, nullptr, W[L][1], RW, NREL2, DIM, DIM, 0);
    edge_scatter_kernel<<<(EDIR * (DIM / 4) + 255) / 256, 256, 0, stream>>>(
        edge_index + EDIR, edge_index + 3 * EDIR, edge_type + EDIR,
        normout, HW, RW, agg, EDIR);

    // bias + 1/3 scale + batchnorm stats, then normalize + tanh
    zero_f32_kernel<<<2, 256, 0, stream>>>(colsum, 512);   // colsum & colsq adjacent
    post_stats_kernel<<<(NENT + 127) / 128, 256, 0, stream>>>(
        agg, lrw, bias[L], colsum, colsq, NENT);
    bn_finalize_kernel<<<1, 256, 0, stream>>>(colsum, colsq, mu, rstd, (float)NENT);
    bn_tanh_kernel<<<(unsigned)(((size_t)NENT * DIM + 255) / 256), 256, 0, stream>>>(
        agg, mu, rstd, xout, (size_t)NENT * DIM);

    // relation path: r_out = rel @ w_rel  (== (rel_all @ w_rel)[:-1])
    gemm(rin, nullptr, W[L][3], rout, NREL2, DIM, DIM, 0);

    xin = xout; rin = rout;
  }

  gather_kernel<<<(NBATCH * DIM + 255) / 256, 256, 0, stream>>>(
      sub, rel, x2, r2, sub_emb, rel_emb, NBATCH);
}
